// Corrector_82841329205652
// MI455X (gfx1250) — matrix operand order repaired and checked by reading, not compiled, not run on hardware
//
#include <hip/hip_runtime.h>
#include <hip/hip_bf16.h>

#define B_ 64
#define S_ 16
#define D_ 128
#define F_ 512
#define H_ 512
#define N_ 4096
#define EPS_ATTN 1e-8f
#define EPS_LN 1e-3f
#define ATTN_SCALE 0.04419417382415922f   // 512^-0.5

typedef __bf16 bf16_t;
typedef __attribute__((ext_vector_type(16))) __bf16 v16bf;
typedef __attribute__((ext_vector_type(8)))  __bf16 v8bf;
typedef __attribute__((ext_vector_type(8)))  float  v8f;

__device__ __forceinline__ v8f vzero8() {
  v8f v;
#pragma unroll
  for (int i = 0; i < 8; ++i) v[i] = 0.0f;
  return v;
}

__device__ __forceinline__ bf16_t f2bf(float f) {
  unsigned u = __float_as_uint(f);
  u += 0x7FFFu + ((u >> 16) & 1u);          // round-to-nearest-even
  unsigned short h = (unsigned short)(u >> 16);
  return __builtin_bit_cast(bf16_t, h);
}

__device__ __forceinline__ float wave_sum32(float v) {
#pragma unroll
  for (int m = 16; m >= 1; m >>= 1) v += __shfl_xor(v, m, 32);
  return v;
}

// A operand 16x32 bf16 tile from row-major [rows][ld] buffer.
// Layout: lanes 0-15 row M=lane, elems 0..15 -> K 0..15;
// lanes 16-31 same rows, K 16..31 (matches load_b16).
__device__ __forceinline__ v16bf load_a16(const bf16_t* base, int ld, int row0, int k0, int lane) {
  int r = row0 + (lane & 15);
  int khalf = (lane >> 4) << 4;
  const bf16_t* p = base + r * ld + k0 + khalf;
  v8bf lo = *(const v8bf*)(p);
  v8bf hi = *(const v8bf*)(p + 8);
  v16bf a;
#pragma unroll
  for (int i = 0; i < 8; ++i) { a[i] = lo[i]; a[i + 8] = hi[i]; }
  return a;
}

// B operand 32x16 bf16 tile; source laid out [cols][ld] (K contiguous per column).
// Layout used here (same k order as load_a16): lanes 0-15 col=lane hold K=0..15, lanes 16-31 hold K=16..31.
__device__ __forceinline__ v16bf load_b16(const bf16_t* base, int ld, int col0, int k0, int lane) {
  int c = col0 + (lane & 15);
  int k = k0 + ((lane >> 4) << 4);
  return *(const v16bf*)(base + c * ld + k);
}

__device__ __forceinline__ v8f wmma_bf16(v16bf a, v16bf b, v8f c) {
  return __builtin_amdgcn_wmma_f32_16x16x32_bf16(false, a, false, b, (short)0, c, false, false);
}

// ---------------- prep kernels ----------------
__global__ void transpose_bf16_kernel(const float* __restrict__ src, bf16_t* __restrict__ dst,
                                      int R, int C) {
  int idx = blockIdx.x * blockDim.x + threadIdx.x;
  if (idx < R * C) {
    int r = idx / C, c = idx % C;
    dst[c * R + r] = f2bf(src[idx]);
  }
}

__global__ void convert_bf16_kernel(const float* __restrict__ src, bf16_t* __restrict__ dst, int n) {
  int idx = blockIdx.x * blockDim.x + threadIdx.x;
  if (idx < n) dst[idx] = f2bf(src[idx]);
}

// ---------------- LN + k/v projection ----------------
#define K1_ROWS 32
#define K1_LDA 520   // bf16 elems; 1040B row stride -> 4-bank rotation, 16B aligned

__global__ __launch_bounds__(256) void kv_kernel(
    const float* __restrict__ feats, const float* __restrict__ ln_g, const float* __restrict__ ln_b,
    const bf16_t* __restrict__ Wkt, const float* __restrict__ bk,
    const bf16_t* __restrict__ Wvt, const float* __restrict__ bv,
    bf16_t* __restrict__ k_out,   // [B*N][128]
    bf16_t* __restrict__ v_out)   // [B][128][N]
{
  __shared__ __align__(16) bf16_t lds[K1_ROWS * K1_LDA];
  const int tid = threadIdx.x, lane = tid & 31, wid = tid >> 5;
  const int gbase = blockIdx.x * K1_ROWS;

  // LayerNorm: 4 rows per wave, 16 f32 values per lane per row
  for (int rr = 0; rr < 4; ++rr) {
    int r = wid * 4 + rr;
    const float* src = feats + (size_t)(gbase + r) * F_;
    float x[16]; float s = 0.f, sq = 0.f;
#pragma unroll
    for (int q = 0; q < 16; ++q) { x[q] = src[lane + 32 * q]; s += x[q]; sq += x[q] * x[q]; }
    s = wave_sum32(s); sq = wave_sum32(sq);
    float mu = s * (1.f / 512.f);
    float var = sq * (1.f / 512.f) - mu * mu;
    float rs = rsqrtf(var + EPS_LN);
#pragma unroll
    for (int q = 0; q < 16; ++q) {
      int c = lane + 32 * q;
      lds[r * K1_LDA + c] = f2bf((x[q] - mu) * rs * ln_g[c] + ln_b[c]);
    }
  }
  __syncthreads();

  // 8 waves: wave -> (proj, mtile, 4 ntiles)
  const int proj = wid >> 2;            // 0 = k, 1 = v
  const int mtile = (wid >> 1) & 1;
  const int nt0 = (wid & 1) * 4;
  const bf16_t* Wt = proj ? Wvt : Wkt;  // [out 128][in 512] bf16
  const float* bias = proj ? bv : bk;

  v8f acc[4];
#pragma unroll
  for (int t = 0; t < 4; ++t) acc[t] = vzero8();

  for (int kk = 0; kk < F_; kk += 32) {
    v16bf a = load_a16(lds, K1_LDA, mtile * 16, kk, lane);
#pragma unroll
    for (int t = 0; t < 4; ++t)
      acc[t] = wmma_bf16(a, load_b16(Wt, F_, (nt0 + t) * 16, kk, lane), acc[t]);
  }

#pragma unroll
  for (int t = 0; t < 4; ++t) {
    int j = (nt0 + t) * 16 + (lane & 15);
    float bj = bias[j];
#pragma unroll
    for (int r = 0; r < 8; ++r) {
      int m = mtile * 16 + r + ((lane >> 4) << 3);
      size_t g = (size_t)(gbase + m);
      bf16_t val = f2bf(acc[t][r] + bj);
      if (proj == 0) {
        k_out[g * D_ + j] = val;
      } else {
        size_t bb = g >> 12, nn = g & 4095;
        v_out[(bb * D_ + j) * (size_t)N_ + nn] = val;   // transposed store
      }
    }
  }
}

// ---------------- q projection (+ zero accumulators) ----------------
__global__ __launch_bounds__(256) void q_kernel(
    const int* __restrict__ is_first, int iter,
    const float* __restrict__ slots,                      // [B][16][128]
    const float* __restrict__ ln_g, const float* __restrict__ ln_b,
    const bf16_t* __restrict__ Wqt, const float* __restrict__ bq,
    bf16_t* __restrict__ q_g, float* __restrict__ U, float* __restrict__ RS)
{
  if (iter >= ((is_first[0] != 0) ? 3 : 2)) return;
  __shared__ __align__(32) char smem[12288];
  float* sl = (float*)smem;                 // 8KB
  bf16_t* xbf = (bf16_t*)(smem + 8192);     // 4KB
  const int tid = threadIdx.x, lane = tid & 31, wid = tid >> 5;
  const int b = blockIdx.x;

  for (int e = tid; e < 2048; e += 256) sl[e] = slots[(size_t)b * 2048 + e];
  for (int e = tid; e < 2048; e += 256) U[(size_t)b * 2048 + e] = 0.f;
  if (tid < 16) RS[b * 16 + tid] = 0.f;
  __syncthreads();

  // LN over 128: 2 rows per wave
  for (int rr = 0; rr < 2; ++rr) {
    int r = wid * 2 + rr;
    float x[4]; float s = 0.f, sq = 0.f;
#pragma unroll
    for (int qi = 0; qi < 4; ++qi) { x[qi] = sl[r * 128 + lane + 32 * qi]; s += x[qi]; sq += x[qi] * x[qi]; }
    s = wave_sum32(s); sq = wave_sum32(sq);
    float mu = s * (1.f / 128.f);
    float var = sq * (1.f / 128.f) - mu * mu;
    float rs = rsqrtf(var + EPS_LN);
#pragma unroll
    for (int qi = 0; qi < 4; ++qi) {
      int c = lane + 32 * qi;
      xbf[r * 128 + c] = f2bf((x[qi] - mu) * rs * ln_g[c] + ln_b[c]);
    }
  }
  __syncthreads();

  // q = LN(slots) @ Wq + bq (SCALE folded in); wave w -> out tile w
  v8f acc = vzero8();
  for (int kk = 0; kk < 128; kk += 32) {
    v16bf a = load_a16(xbf, 128, 0, kk, lane);
    acc = wmma_bf16(a, load_b16(Wqt, 128, wid * 16, kk, lane), acc);
  }
  int j = wid * 16 + (lane & 15);
  float bj = bq[j];
  bf16_t* qdst = q_g + (size_t)b * 2048;
#pragma unroll
  for (int r = 0; r < 8; ++r) {
    int s = r + ((lane >> 4) << 3);
    qdst[s * 128 + j] = f2bf((acc[r] + bj) * ATTN_SCALE);
  }
}

// ---------------- attention partials: dots -> softmax(slots) -> partial U/rowsum ----------------
__global__ __launch_bounds__(256) void attn_kernel(
    const int* __restrict__ is_first, int iter,
    const bf16_t* __restrict__ q_g,   // [B][16][128]
    const bf16_t* __restrict__ k_g,   // [B*N][128]
    const bf16_t* __restrict__ v_t,   // [B][128][N]
    float* __restrict__ U,            // [B][16][128]
    float* __restrict__ RS)           // [B][16]
{
  if (iter >= ((is_first[0] != 0) ? 3 : 2)) return;
  __shared__ __align__(32) char smem[16384];
  bf16_t* qbf = (bf16_t*)smem;              // 4KB
  float* dots = (float*)(smem + 4096);      // 8KB [16 slots][128 pixels]
  bf16_t* attnbf = (bf16_t*)(smem + 12288); // 4KB

  const int tid = threadIdx.x, lane = tid & 31, wid = tid >> 5;
  const int b = blockIdx.x >> 5;            // 32 chunks of 128 pixels per batch
  const int pb = (blockIdx.x & 31) * 128;
  const bf16_t* kb = k_g + ((size_t)b * N_ + pb) * D_;
  const bf16_t* vb = v_t + (size_t)b * D_ * (size_t)N_;

  {
    const bf16_t* qsrc = q_g + (size_t)b * 2048;
    for (int e = tid; e < 2048; e += 256) qbf[e] = qsrc[e];
  }
  __syncthreads();

  // dots: wave w -> pixel tile w (16 pixels), K=128 feature dims
  {
    v8f acc = vzero8();
    const bf16_t* kbase = kb + (size_t)(wid * 16) * D_;
    for (int kk = 0; kk < D_; kk += 32) {
      v16bf a = load_a16(qbf, D_, 0, kk, lane);
      acc = wmma_bf16(a, load_b16(kbase, D_, 0, kk, lane), acc);
    }
    int c = wid * 16 + (lane & 15);
#pragma unroll
    for (int r = 0; r < 8; ++r) dots[(r + ((lane >> 4) << 3)) * 128 + c] = acc[r];
  }
  __syncthreads();

  // per-pixel softmax over 16 slots (threads 0..127, one pixel each)
  float rsum[16];
#pragma unroll
  for (int s = 0; s < 16; ++s) rsum[s] = 0.f;
  if (tid < 128) {
    float vv[16]; float mx = -1e30f;
#pragma unroll
    for (int s = 0; s < 16; ++s) { vv[s] = dots[s * 128 + tid]; mx = fmaxf(mx, vv[s]); }
    float sm = 0.f;
#pragma unroll
    for (int s = 0; s < 16; ++s) { vv[s] = __expf(vv[s] - mx); sm += vv[s]; }
    float inv = 1.f / sm;
#pragma unroll
    for (int s = 0; s < 16; ++s) {
      float a = vv[s] * inv + EPS_ATTN;
      rsum[s] += a;
      attnbf[s * 128 + tid] = f2bf(a);
    }
  }
  __syncthreads();

  // partial U tile: wave w -> dim tile w, K = 128 pixels in this chunk
  {
    v8f acc = vzero8();
    int d = wid * 16 + (lane & 15);
    for (int kk = 0; kk < 128; kk += 32) {
      v16bf a = load_a16(attnbf, 128, 0, kk, lane);
      v16bf bm = *(const v16bf*)(vb + (size_t)d * N_ + pb + kk + ((lane >> 4) << 4));
      acc = wmma_bf16(a, bm, acc);
    }
    float* Ub = U + (size_t)b * 2048;
#pragma unroll
    for (int r = 0; r < 8; ++r) {
      int s = r + ((lane >> 4) << 3);
      atomicAdd(&Ub[s * D_ + d], acc[r]);
    }
  }
  __syncthreads();

  // reduce per-chunk rowsum and atomically add
  if (tid < 128) {
#pragma unroll
    for (int s = 0; s < 16; ++s) dots[s * 128 + tid] = rsum[s];
  }
  __syncthreads();
#pragma unroll
  for (int rr = 0; rr < 2; ++rr) {
    int s = wid * 2 + rr;
    float v = dots[s * 128 + lane] + dots[s * 128 + lane + 32] +
              dots[s * 128 + lane + 64] + dots[s * 128 + lane + 96];
    v = wave_sum32(v);
    if (lane == 0) atomicAdd(&RS[b * 16 + s], v);
  }
}

// ---------------- GRU + MLP update ----------------
__global__ __launch_bounds__(256) void update_kernel(
    const int* __restrict__ is_first, int iter,
    const float* __restrict__ slots_prev,
    const float* __restrict__ U, const float* __restrict__ RS,
    const bf16_t* __restrict__ Wih, const bf16_t* __restrict__ Whh,
    const float* __restrict__ b_ih, const float* __restrict__ b_hh,
    const float* __restrict__ ln_g, const float* __restrict__ ln_b,
    const bf16_t* __restrict__ W1t, const float* __restrict__ b1,
    const bf16_t* __restrict__ W2t, const float* __restrict__ b2,
    float* __restrict__ slots_out, float* __restrict__ final_out)
{
  if (iter >= ((is_first[0] != 0) ? 3 : 2)) return;
  __shared__ __align__(32) char smem[49152];
  float* sl = (float*)smem;                 // 8KB slots
  bf16_t* xbf = (bf16_t*)(smem + 8192);     // 4KB updates / LN
  bf16_t* hbf = (bf16_t*)(smem + 12288);    // 4KB h_prev bf16
  float* G = (float*)(smem + 16384);        // 24KB [16][384] gates
  float* Hb = (float*)(smem + 40960);       // 8KB  [16][128] h_n part
  bf16_t* hidbf = (bf16_t*)(smem + 16384);  // 16KB, aliases G (after gates)

  const int tid = threadIdx.x, lane = tid & 31, wid = tid >> 5;
  const int b = blockIdx.x;

  for (int e = tid; e < 2048; e += 256) {
    float h = slots_prev[(size_t)b * 2048 + e];
    sl[e] = h;
    hbf[e] = f2bf(h);
  }
  {
    const float* Ub = U + (size_t)b * 2048;
    const float* rs = RS + b * 16;
    for (int e = tid; e < 2048; e += 256) xbf[e] = f2bf(Ub[e] / rs[e >> 7]);
  }
  __syncthreads();

  // gi = x @ W_ih^T, gh = h @ W_hh^T : 24 out-tiles, wave w -> tiles w, w+8, w+16
  for (int tt = 0; tt < 3; ++tt) {
    int jt = wid + tt * 8;
    v8f gi = vzero8(), gh = vzero8();
    for (int kk = 0; kk < 128; kk += 32) {
      v16bf ax = load_a16(xbf, 128, 0, kk, lane);
      v16bf ah = load_a16(hbf, 128, 0, kk, lane);
      gi = wmma_bf16(ax, load_b16(Wih, 128, jt * 16, kk, lane), gi);
      gh = wmma_bf16(ah, load_b16(Whh, 128, jt * 16, kk, lane), gh);
    }
    int j = jt * 16 + (lane & 15);
    float bi = b_ih[j], bh = b_hh[j];
#pragma unroll
    for (int r = 0; r < 8; ++r) {
      int s = r + ((lane >> 4) << 3);
      if (jt < 16) {                    // r,z parts: sum immediately
        G[s * 384 + j] = gi[r] + gh[r] + bi + bh;
      } else {                          // n part: keep i_n and h_n separate
        G[s * 384 + j] = gi[r] + bi;
        Hb[s * 128 + (j - 256)] = gh[r] + bh;
      }
    }
  }
  __syncthreads();

  for (int e = tid; e < 2048; e += 256) {
    int s = e >> 7, d = e & 127;
    float r = 1.f / (1.f + __expf(-G[s * 384 + d]));
    float z = 1.f / (1.f + __expf(-G[s * 384 + 128 + d]));
    float n = tanhf(G[s * 384 + 256 + d] + r * Hb[s * 128 + d]);
    sl[e] = (1.f - z) * n + z * sl[e];
  }
  __syncthreads();

  // LN_mlp(slots) -> xbf
  for (int rr = 0; rr < 2; ++rr) {
    int r = wid * 2 + rr;
    float x[4]; float s = 0.f, sq = 0.f;
#pragma unroll
    for (int qi = 0; qi < 4; ++qi) { x[qi] = sl[r * 128 + lane + 32 * qi]; s += x[qi]; sq += x[qi] * x[qi]; }
    s = wave_sum32(s); sq = wave_sum32(sq);
    float mu = s * (1.f / 128.f);
    float var = sq * (1.f / 128.f) - mu * mu;
    float rsn = rsqrtf(var + EPS_LN);
#pragma unroll
    for (int qi = 0; qi < 4; ++qi) {
      int c = lane + 32 * qi;
      xbf[r * 128 + c] = f2bf((x[qi] - mu) * rsn * ln_g[c] + ln_b[c]);
    }
  }
  __syncthreads();

  // hid = relu(x @ W1 + b1): 32 out-tiles, wave w -> w, w+8, w+16, w+24
  for (int tt = 0; tt < 4; ++tt) {
    int jt = wid + tt * 8;
    v8f acc = vzero8();
    for (int kk = 0; kk < 128; kk += 32) {
      v16bf a = load_a16(xbf, 128, 0, kk, lane);
      acc = wmma_bf16(a, load_b16(W1t, 128, jt * 16, kk, lane), acc);
    }
    int j = jt * 16 + (lane & 15);
    float bj = b1[j];
#pragma unroll
    for (int r = 0; r < 8; ++r) {
      int s = r + ((lane >> 4) << 3);
      hidbf[s * 512 + j] = f2bf(fmaxf(acc[r] + bj, 0.f));
    }
  }
  __syncthreads();

  // out = hid @ W2 + b2; slots += out
  {
    v8f acc = vzero8();
    for (int kk = 0; kk < 512; kk += 32) {
      v16bf a = load_a16(hidbf, 512, 0, kk, lane);
      acc = wmma_bf16(a, load_b16(W2t, 512, wid * 16, kk, lane), acc);
    }
    int j = wid * 16 + (lane & 15);
    float bj = b2[j];
#pragma unroll
    for (int r = 0; r < 8; ++r) {
      int s = r + ((lane >> 4) << 3);
      float v = sl[s * 128 + j] + acc[r] + bj;
      slots_out[(size_t)b * 2048 + s * 128 + j] = v;
      final_out[(size_t)b * 2048 + s * 128 + j] = v;
    }
  }
}

// ---------------- host ----------------
extern "C" void kernel_launch(void* const* d_in, const int* in_sizes, int n_in,
                              void* d_out, int out_size, void* d_ws, size_t ws_size,
                              hipStream_t stream) {
  const float* feats     = (const float*)d_in[0];
  const float* slots     = (const float*)d_in[1];
  const int*   is_first  = (const int*)d_in[2];
  const float* ln_in_g   = (const float*)d_in[3];
  const float* ln_in_b   = (const float*)d_in[4];
  const float* ln_slot_g = (const float*)d_in[5];
  const float* ln_slot_b = (const float*)d_in[6];
  const float* ln_mlp_g  = (const float*)d_in[7];
  const float* ln_mlp_b  = (const float*)d_in[8];
  const float* Wq        = (const float*)d_in[9];
  const float* bq        = (const float*)d_in[10];
  const float* Wk        = (const float*)d_in[11];
  const float* bk        = (const float*)d_in[12];
  const float* Wv        = (const float*)d_in[13];
  const float* bv        = (const float*)d_in[14];
  const float* W_ih      = (const float*)d_in[15];
  const float* W_hh      = (const float*)d_in[16];
  const float* b_ih      = (const float*)d_in[17];
  const float* b_hh      = (const float*)d_in[18];
  const float* W1        = (const float*)d_in[19];
  const float* b1        = (const float*)d_in[20];
  const float* W2        = (const float*)d_in[21];
  const float* b2        = (const float*)d_in[22];
  float* outp = (float*)d_out;

  char* ws = (char*)d_ws;
  size_t off = 0;
  auto alloc = [&](size_t bytes) { char* p = ws + off; off = (off + bytes + 255) & ~(size_t)255; return p; };
  bf16_t* k_bf  = (bf16_t*)alloc((size_t)B_ * N_ * D_ * 2);     // 64 MB
  bf16_t* v_tb  = (bf16_t*)alloc((size_t)B_ * N_ * D_ * 2);     // 64 MB
  bf16_t* q_g   = (bf16_t*)alloc((size_t)B_ * S_ * D_ * 2);
  float*  U     = (float*)alloc((size_t)B_ * S_ * D_ * 4);
  float*  RS    = (float*)alloc((size_t)B_ * S_ * 4);
  float*  wslot = (float*)alloc((size_t)B_ * S_ * D_ * 4);
  bf16_t* Wqt   = (bf16_t*)alloc((size_t)D_ * D_ * 2);
  bf16_t* Wkt   = (bf16_t*)alloc((size_t)F_ * D_ * 2);
  bf16_t* Wvt   = (bf16_t*)alloc((size_t)F_ * D_ * 2);
  bf16_t* W1t   = (bf16_t*)alloc((size_t)D_ * H_ * 2);
  bf16_t* W2t   = (bf16_t*)alloc((size_t)H_ * D_ * 2);
  bf16_t* Wihb  = (bf16_t*)alloc((size_t)3 * D_ * D_ * 2);
  bf16_t* Whhb  = (bf16_t*)alloc((size_t)3 * D_ * D_ * 2);
  (void)ws_size; (void)in_sizes; (void)n_in; (void)out_size;

  // weight prep: [in,out] -> bf16 [out][in]; GRU weights already [out][in]
  transpose_bf16_kernel<<<(D_ * D_ + 255) / 256, 256, 0, stream>>>(Wq, Wqt, D_, D_);
  transpose_bf16_kernel<<<(F_ * D_ + 255) / 256, 256, 0, stream>>>(Wk, Wkt, F_, D_);
  transpose_bf16_kernel<<<(F_ * D_ + 255) / 256, 256, 0, stream>>>(Wv, Wvt, F_, D_);
  transpose_bf16_kernel<<<(D_ * H_ + 255) / 256, 256, 0, stream>>>(W1, W1t, D_, H_);
  transpose_bf16_kernel<<<(H_ * D_ + 255) / 256, 256, 0, stream>>>(W2, W2t, H_, D_);
  convert_bf16_kernel<<<(3 * D_ * D_ + 255) / 256, 256, 0, stream>>>(W_ih, Wihb, 3 * D_ * D_);
  convert_bf16_kernel<<<(3 * D_ * D_ + 255) / 256, 256, 0, stream>>>(W_hh, Whhb, 3 * D_ * D_);

  // fused LN + k/v projection (single pass over 512MB of features)
  kv_kernel<<<(B_ * N_) / K1_ROWS, 256, 0, stream>>>(feats, ln_in_g, ln_in_b,
                                                     Wkt, bk, Wvt, bv, k_bf, v_tb);

  // up to 3 iterations; each kernel self-gates on device-side is_first
  for (int it = 0; it < 3; ++it) {
    const float* sp = (it == 0) ? slots : wslot;
    q_kernel<<<B_, 256, 0, stream>>>(is_first, it, sp, ln_slot_g, ln_slot_b, Wqt, bq,
                                     q_g, U, RS);
    attn_kernel<<<B_ * 32, 256, 0, stream>>>(is_first, it, q_g, k_bf, v_tb, U, RS);
    update_kernel<<<B_, 256, 0, stream>>>(is_first, it, sp, U, RS, Wihb, Whhb, b_ih, b_hh,
                                          ln_mlp_g, ln_mlp_b, W1t, b1, W2t, b2,
                                          wslot, outp);
  }
}
